// RoPEAttentionHead_6700148981847
// MI455X (gfx1250) — matrix operand order repaired and checked by reading, not compiled, not run on hardware
//
#include <hip/hip_runtime.h>
#include <hip/hip_bf16.h>
#include <stdint.h>

#define B_  8
#define M_  2048
#define D_  128
#define TS  128    // row tile per block
#define NW  8      // waves per 256-thread block

typedef __attribute__((ext_vector_type(16))) __bf16 v16bf;
typedef __attribute__((ext_vector_type(8)))  float  v8f;
typedef __attribute__((ext_vector_type(4)))  unsigned int u32x4;
typedef __attribute__((ext_vector_type(4)))  int    i32x4;
typedef __attribute__((ext_vector_type(8)))  int    i32x8;

// ---------- helpers ----------------------------------------------------------

__device__ __forceinline__ unsigned short f32_to_bf16_bits(float f) {
  unsigned u = __builtin_bit_cast(unsigned, f);
  u += 0x7FFFu + ((u >> 16) & 1u);               // round-to-nearest-even
  return (unsigned short)(u >> 16);
}
__device__ __forceinline__ __bf16 bits_to_bf16(unsigned short s) {
  return __builtin_bit_cast(__bf16, s);
}
__device__ __forceinline__ v8f wmma_bf16(v16bf a, v16bf b, v8f c) {
  // D = A(16x32) * B(32x16) + C, f32 accumulate
  return __builtin_amdgcn_wmma_f32_16x16x32_bf16(
      false, a, false, b, (short)0, c, false, false);
}
__device__ __forceinline__ unsigned int lds_off(const void* p) {
  return (unsigned int)(uintptr_t)p;   // low 32 bits of flat LDS addr = offset
}

// ---------- Tensor Data Mover: generic 2-D tile -> LDS -----------------------

#if defined(__has_builtin)
#if __has_builtin(__builtin_amdgcn_tensor_load_to_lds) && \
    __has_builtin(__builtin_amdgcn_s_wait_tensorcnt)
#define HAVE_TDM 1
#endif
#endif
#ifndef HAVE_TDM
#define HAVE_TDM 0
#endif

#if HAVE_TDM
// dsz_code: 0=1B,1=2B,2=4B,3=8B. dims/strides in elements of that size.
__device__ __forceinline__ void tdm_load_2d(const void* gbase,
                                            unsigned int ldsoff,
                                            unsigned int dsz_code,
                                            unsigned int dim0, unsigned int dim1,
                                            unsigned int tile0, unsigned int tile1,
                                            unsigned int stride0) {
  unsigned long long ga = (unsigned long long)(uintptr_t)gbase;
  u32x4 g0;
  g0[0] = 1u;                                           // count=1, user desc
  g0[1] = ldsoff;                                       // lds_addr (bytes)
  g0[2] = (unsigned int)(ga & 0xFFFFFFFFull);           // global_addr[31:0]
  g0[3] = (unsigned int)((ga >> 32) & 0x01FFFFFFull)    // global_addr[56:32]
          | (2u << 30);                                 // type=2 ("image")
  i32x8 g1;
  g1[0] = (int)(dsz_code << 16);                 // mask=0 | data_size
  g1[1] = (int)((dim0 & 0xFFFFu) << 16);         // tensor_dim0 lo -> bits 63:48
  g1[2] = (int)((dim0 >> 16) | ((dim1 & 0xFFFFu) << 16));
  g1[3] = (int)((dim1 >> 16) | (tile0 << 16));   // | tile_dim0
  g1[4] = (int)(tile1);                          // tile_dim1 | tile_dim2=0
  g1[5] = (int)(stride0);                        // tensor_dim0_stride lo
  g1[6] = 0;
  g1[7] = 0;
  i32x4 gz = (i32x4)0;
#if __clang_major__ >= 23
  i32x8 gz8 = (i32x8)0;
  __builtin_amdgcn_tensor_load_to_lds(g0, g1, gz, gz, gz8, 0);
#else
  __builtin_amdgcn_tensor_load_to_lds(g0, g1, gz, gz, 0);
#endif
}
#endif

// ---------- Kernel 1: fused QKV projection + exact RoPE ----------------------
// grid = B*(M/TS) blocks of 256 threads; wave w owns rows [tile*128+16w, +16)
// LDS: weights bf16 96KB | x tile f32 64KB  (x tile arrives via TDM)

__global__ void qkv_rope_kernel(const float* __restrict__ x,
                                const float* __restrict__ wq,
                                const float* __restrict__ wk,
                                const float* __restrict__ wv,
                                const float* __restrict__ R,
                                unsigned short* __restrict__ qws,
                                unsigned short* __restrict__ kws,
                                unsigned short* __restrict__ vws) {
  extern __shared__ __align__(16) unsigned char smemA[];
  unsigned short* wlds = (unsigned short*)smemA;                 // [3][128][128]
  float* xlds = (float*)(smemA + (size_t)3 * D_ * D_ * sizeof(unsigned short));

  const int tid  = threadIdx.x;
  const int wave = tid >> 5;
  const int lane = tid & 31;
  const int half = lane >> 4;
  const int ln16 = lane & 15;
  const int b    = blockIdx.x >> 4;
  const int row0 = (blockIdx.x & 15) * TS;

  const float* xbase = x + ((size_t)b * M_ + row0) * D_;
#if HAVE_TDM
  if (wave == 0)   // DMA the f32 x tile while everyone stages weights
    tdm_load_2d(xbase, lds_off(xlds), /*4B*/2, D_, M_, D_, TS, D_);
#endif

  // stage all three weight matrices into LDS as bf16 (row = out feat e, col = d)
  for (int idx = tid; idx < 3 * D_ * D_; idx += blockDim.x) {
    int mat = idx >> 14;
    int off = idx & 16383;
    const float* w = (mat == 0) ? wq : (mat == 1 ? wk : wv);
    wlds[idx] = f32_to_bf16_bits(w[off]);
  }

#if HAVE_TDM
  __builtin_amdgcn_s_wait_tensorcnt(0);
#else
  for (int i = tid; i < TS * D_ / 4; i += blockDim.x)
    ((float4*)xlds)[i] = ((const float4*)xbase)[i];
#endif
  __syncthreads();

  // A fragments for this wave's 16 rows of x (K = 128 -> 4 fragments of K=32)
  const int lrow = wave * 16 + ln16;          // A layout: row = lane%16
  v16bf afr[4];
#pragma unroll
  for (int ks = 0; ks < 4; ++ks) {
#pragma unroll
    for (int e = 0; e < 16; ++e) {
      int k = ks * 32 + (e & 7) + half * 8 + ((e >> 3) << 4);
      afr[ks][e] = bits_to_bf16(f32_to_bf16_bits(xlds[lrow * D_ + k]));
    }
  }

  for (int mat = 0; mat < 3; ++mat) {
    v8f acc[8];
#pragma unroll
    for (int t = 0; t < 8; ++t)
#pragma unroll
      for (int r = 0; r < 8; ++r) acc[t][r] = 0.0f;

#pragma unroll
    for (int t = 0; t < 8; ++t) {
#pragma unroll
      for (int ks = 0; ks < 4; ++ks) {
        v16bf bfr;                       // B[k][n] = W[n][k]
#pragma unroll
        for (int e = 0; e < 16; ++e) {
          int k = ks * 32 + (e & 7) + half * 8 + ((e >> 3) << 4);
          int n = t * 16 + ln16;
          bfr[e] = bits_to_bf16(wlds[mat * 16384 + n * D_ + k]);
        }
        acc[t] = wmma_bf16(afr[ks], bfr, acc[t]);
      }
    }

    // epilogue: RoPE (exact via R's two nonzeros per column) for q,k; store bf16
    unsigned short* dst = (mat == 0) ? qws : (mat == 1 ? kws : vws);
#pragma unroll
    for (int t = 0; t < 8; ++t) {
      int e = t * 16 + ln16;             // C layout: col = lane%16
#pragma unroll
      for (int r = 0; r < 8; ++r) {
        int m = row0 + wave * 16 + half * 8 + r;   // C layout row
        float val = acc[t][r];
        if (mat < 2) {
          float partner = __shfl_xor(val, 1, 32);  // q[e^1] lives in lane^1
          float diag = R[((size_t)m * D_ + e) * D_ + e];
          float offd = R[((size_t)m * D_ + (e ^ 1)) * D_ + e];
          val = val * diag + partner * offd;       // sum_d q[d]*R[d,e]
        }
        dst[((size_t)b * M_ + m) * D_ + e] = f32_to_bf16_bits(val);
      }
    }
  }
}

// ---------- Kernel 2: causal flash attention (double-buffered TDM, WMMA) ----
// grid = B*(M/TS); block handles 128 query rows; wave w owns 16 of them.
// LDS: K[2] 64KB | V[2] 64KB | Q 32KB | P 32KB = 192KB

__global__ void attn_kernel(const unsigned short* __restrict__ qws,
                            const unsigned short* __restrict__ kws,
                            const unsigned short* __restrict__ vws,
                            float* __restrict__ out) {
  extern __shared__ __align__(16) unsigned char smemB[];
  unsigned short* base = (unsigned short*)smemB;
  unsigned short* kbuf = base;                       // 2 x 128x128 bf16
  unsigned short* vbuf = base + 2 * TS * D_;         // 2 x 128x128 bf16
  unsigned short* qlds = base + 4 * TS * D_;         // 128x128 bf16
  unsigned short* plds = base + 5 * TS * D_;         // 8 waves x 16 x 128

  const int tid  = threadIdx.x;
  const int wave = tid >> 5;
  const int lane = tid & 31;
  const int half = lane >> 4;
  const int ln16 = lane & 15;
  const int b    = blockIdx.x >> 4;
  const int qi   = blockIdx.x & 15;
  const int m0   = qi * TS + wave * 16;
  const float scale = 0.08838834764831845f;          // 1/sqrt(128)

  const unsigned short* qg = qws + ((size_t)b * M_ + qi * TS) * D_;

  // ---- prologue: DMA Q tile + first K/V tiles (buffer 0)
#if HAVE_TDM
  if (wave == 0)
    tdm_load_2d(kws + ((size_t)b * M_) * D_, lds_off(kbuf), /*2B*/1,
                D_, M_, D_, TS, D_);
  else if (wave == 1)
    tdm_load_2d(vws + ((size_t)b * M_) * D_, lds_off(vbuf), 1,
                D_, M_, D_, TS, D_);
  else if (wave == 2)
    tdm_load_2d(qg, lds_off(qlds), 1, D_, M_, D_, TS, D_);
  __builtin_amdgcn_s_wait_tensorcnt(0);
#else
  for (int i = tid; i < TS * D_ / 8; i += blockDim.x)
    ((uint4*)qlds)[i] = ((const uint4*)qg)[i];
#endif
  __syncthreads();

  // Q fragments for this wave (from LDS)
  v16bf qfr[4];
#pragma unroll
  for (int ks = 0; ks < 4; ++ks) {
#pragma unroll
    for (int e = 0; e < 16; ++e) {
      int k = ks * 32 + (e & 7) + half * 8 + ((e >> 3) << 4);
      qfr[ks][e] = bits_to_bf16(qlds[(wave * 16 + ln16) * D_ + k]);
    }
  }

  v8f o[8];
  float mrow[8], lrow[8];
#pragma unroll
  for (int t = 0; t < 8; ++t)
#pragma unroll
    for (int r = 0; r < 8; ++r) o[t][r] = 0.0f;
#pragma unroll
  for (int r = 0; r < 8; ++r) { mrow[r] = -1e30f; lrow[r] = 0.0f; }

  for (int j = 0; j <= qi; ++j) {
    const int buf = j & 1;
    unsigned short* kt = kbuf + buf * (TS * D_);
    unsigned short* vt = vbuf + buf * (TS * D_);

#if HAVE_TDM
    if (j < qi) {    // prefetch tile j+1 into the other buffer, wait for tile j
      const int nb = buf ^ 1;
      if (wave == 0)
        tdm_load_2d(kws + ((size_t)b * M_ + (j + 1) * TS) * D_,
                    lds_off(kbuf + nb * (TS * D_)), 1, D_, M_, D_, TS, D_);
      else if (wave == 1)
        tdm_load_2d(vws + ((size_t)b * M_ + (j + 1) * TS) * D_,
                    lds_off(vbuf + nb * (TS * D_)), 1, D_, M_, D_, TS, D_);
      __builtin_amdgcn_s_wait_tensorcnt(1);   // in-order: tile j has landed
    } else {
      __builtin_amdgcn_s_wait_tensorcnt(0);
    }
#else
    {
      const unsigned short* kb = kws + ((size_t)b * M_ + j * TS) * D_;
      const unsigned short* vb = vws + ((size_t)b * M_ + j * TS) * D_;
      for (int i = tid; i < TS * D_ / 8; i += blockDim.x) {
        ((uint4*)kt)[i] = ((const uint4*)kb)[i];
        ((uint4*)vt)[i] = ((const uint4*)vb)[i];
      }
    }
#endif
    __syncthreads();   // tile j visible to all waves

    // S = q_rot . k_rot^T  (16 rows x 128 keys per wave)
    v8f s[8];
#pragma unroll
    for (int t = 0; t < 8; ++t) {
#pragma unroll
      for (int r = 0; r < 8; ++r) s[t][r] = 0.0f;
#pragma unroll
      for (int ks = 0; ks < 4; ++ks) {
        v16bf bfr;                      // B[k=e][n=key] = k_rot[n][e]
#pragma unroll
        for (int e = 0; e < 16; ++e)
          bfr[e] = bits_to_bf16(kt[(t * 16 + ln16) * D_ + ks * 32 + (e & 7) + half * 8 + ((e >> 3) << 4)]);
        s[t] = wmma_bf16(qfr[ks], bfr, s[t]);
      }
    }

    // causal mask + scale
#pragma unroll
    for (int t = 0; t < 8; ++t) {
      int n = j * TS + t * 16 + ln16;
#pragma unroll
      for (int r = 0; r < 8; ++r) {
        int m = m0 + half * 8 + r;
        s[t][r] = (n > m) ? -1e30f : s[t][r] * scale;
      }
    }

    // online softmax (row spread over 16 lanes of one half + 8 n-tiles)
    float tmax[8];
#pragma unroll
    for (int r = 0; r < 8; ++r) {
      float v = s[0][r];
#pragma unroll
      for (int t = 1; t < 8; ++t) v = fmaxf(v, s[t][r]);
      tmax[r] = v;
    }
#pragma unroll
    for (int off = 1; off < 16; off <<= 1)
#pragma unroll
      for (int r = 0; r < 8; ++r)
        tmax[r] = fmaxf(tmax[r], __shfl_xor(tmax[r], off, 32));

#pragma unroll
    for (int r = 0; r < 8; ++r) {
      float mnew  = fmaxf(mrow[r], tmax[r]);
      float alpha = __expf(mrow[r] - mnew);
      mrow[r] = mnew;
      lrow[r] *= alpha;
#pragma unroll
      for (int t = 0; t < 8; ++t) o[t][r] *= alpha;
    }

    float rsum[8];
#pragma unroll
    for (int r = 0; r < 8; ++r) rsum[r] = 0.0f;
#pragma unroll
    for (int t = 0; t < 8; ++t)
#pragma unroll
      for (int r = 0; r < 8; ++r) {
        float p = __expf(s[t][r] - mrow[r]);
        s[t][r] = p;
        rsum[r] += p;
      }
#pragma unroll
    for (int off = 1; off < 16; off <<= 1)
#pragma unroll
      for (int r = 0; r < 8; ++r) rsum[r] += __shfl_xor(rsum[r], off, 32);
#pragma unroll
    for (int r = 0; r < 8; ++r) lrow[r] += rsum[r];

    // re-layout P: C-layout regs -> LDS -> A-layout fragments (same wave, in-order)
#pragma unroll
    for (int t = 0; t < 8; ++t)
#pragma unroll
      for (int r = 0; r < 8; ++r)
        plds[wave * 2048 + (half * 8 + r) * D_ + t * 16 + ln16] =
            f32_to_bf16_bits(s[t][r]);

    v16bf pa[4];
#pragma unroll
    for (int ks = 0; ks < 4; ++ks)
#pragma unroll
      for (int e = 0; e < 16; ++e) {
        int k = ks * 32 + (e & 7) + half * 8 + ((e >> 3) << 4);
        pa[ks][e] = bits_to_bf16(plds[wave * 2048 + ln16 * D_ + k]);
      }

    // O += P . V
#pragma unroll
    for (int te = 0; te < 8; ++te) {
#pragma unroll
      for (int ks = 0; ks < 4; ++ks) {
        v16bf vb;                      // B[k=key][n=e] = v[key][e]
#pragma unroll
        for (int e = 0; e < 16; ++e)
          vb[e] = bits_to_bf16(vt[(ks * 32 + (e & 7) + half * 8 + ((e >> 3) << 4)) * D_ + te * 16 + ln16]);
        o[te] = wmma_bf16(pa[ks], vb, o[te]);
      }
    }

    __syncthreads();   // all waves done with tile j before its buffer is reissued
  }

  // final normalize + store f32
#pragma unroll
  for (int te = 0; te < 8; ++te)
#pragma unroll
    for (int r = 0; r < 8; ++r) {
      int m = m0 + half * 8 + r;
      out[((size_t)b * M_ + m) * D_ + te * 16 + ln16] = o[te][r] / lrow[r];
    }
}

// ---------- host launch ------------------------------------------------------

extern "C" void kernel_launch(void* const* d_in, const int* in_sizes, int n_in,
                              void* d_out, int out_size, void* d_ws, size_t ws_size,
                              hipStream_t stream) {
  const float* x  = (const float*)d_in[0];
  const float* wq = (const float*)d_in[1];
  const float* wk = (const float*)d_in[2];
  const float* wv = (const float*)d_in[3];
  const float* R  = (const float*)d_in[4];
  float* out = (float*)d_out;

  unsigned short* qws = (unsigned short*)d_ws;              // bf16 q_rot
  unsigned short* kws = qws + (size_t)B_ * M_ * D_;         // bf16 k_rot
  unsigned short* vws = kws + (size_t)B_ * M_ * D_;         // bf16 v

  dim3 grid(B_ * (M_ / TS));
  dim3 block(256);
  // Kernel A: 96KB bf16 weights + 64KB f32 x tile = 160KB
  size_t smemA = (size_t)3 * D_ * D_ * sizeof(unsigned short)
               + (size_t)TS * D_ * sizeof(float);
  // Kernel B: 2xK + 2xV + Q + P tiles (bf16) = 192KB
  size_t smemB = (size_t)6 * TS * D_ * sizeof(unsigned short);

  (void)hipFuncSetAttribute((const void*)qkv_rope_kernel,
                            hipFuncAttributeMaxDynamicSharedMemorySize,
                            (int)smemA);
  (void)hipFuncSetAttribute((const void*)attn_kernel,
                            hipFuncAttributeMaxDynamicSharedMemorySize,
                            (int)smemB);

  qkv_rope_kernel<<<grid, block, smemA, stream>>>(x, wq, wk, wv, R, qws, kws, vws);
  attn_kernel<<<grid, block, smemB, stream>>>(qws, kws, vws, out);
}
